// MultilayerGRU_87033217286978
// MI455X (gfx1250) — compile-verified, matrix operand order checked by reading, not run on hardware
//
#include <hip/hip_runtime.h>
#include <hip/hip_bf16.h>

// ---------------------------------------------------------------------------
// Multilayer GRU for MI455X (gfx1250), wave32 + WMMA f32_16x16x32_f16.
//
// Phase A: pack x -> f16 (row = t*32+b), pack all weights into WMMA B-operand
//          per-lane layout (f16).
// Per layer: chunked over time (CT=256 steps):
//   Phase B1: Xp[chunk] = seq_in @ [Wxz|Wxr|Wxg] + bias   (big WMMA GEMM, f32 out)
//   Phase B2: persistent 8-WG recurrent kernel, 2 grid barriers per step:
//             stage1: z,r = sigmoid(Xp + h@Wh{z,r}); rh = r*h  (rh -> global)
//             stage2: g = tanh(Xp_g + rh@Whg); h = z*h+(1-z)*g
// Phase C: layer_output = seq_l2 @ Wo + bo (WMMA GEMM, scattered f32 store).
// ---------------------------------------------------------------------------

typedef _Float16 h16;
typedef __attribute__((ext_vector_type(16))) _Float16 v16h;
typedef __attribute__((ext_vector_type(8)))  _Float16 v8h;
typedef __attribute__((ext_vector_type(8)))  float    v8f;

#define B_ 32
#define S_ 2048
#define I_ 256
#define H_ 512
#define O_ 256
#define L_ 3
#define CT 256        // time-steps per chunk
#define NWG_REC 8     // workgroups in the persistent recurrent kernel

// D = A(16x32 f16) * B(32x16 f16) + C(16x16 f32)
__device__ __forceinline__ v8f wmma16(v16h a, v16h b, v8f c) {
  return __builtin_amdgcn_wmma_f32_16x16x32_f16(false, a, false, b, (short)0, c,
                                                false, false);
}

// A-operand gather for one 32-K block from a row-major f16 row pointer.
// ISA 16-bit A layout: lane l (row = l&15) holds K = kb8+{0..7} then kb8+{16..23},
// kb8 = (l>>4)*8  ->  two contiguous 8-half (16B) chunks.
__device__ __forceinline__ v16h load_a(const h16* __restrict__ rowp, int kb, int kb8) {
  v8h lo = *(const v8h*)(rowp + kb * 32 + kb8);
  v8h hi = *(const v8h*)(rowp + kb * 32 + kb8 + 16);
  return __builtin_shufflevector(lo, hi, 0, 1, 2, 3, 4, 5, 6, 7,
                                 8, 9, 10, 11, 12, 13, 14, 15);
}

// ---------------------------------------------------------------------------
// Packing kernels
// ---------------------------------------------------------------------------

// x (B,S,I) f32 -> xbf[(t*32+b)*I + i] f16
__global__ void pack_x_kernel(h16* __restrict__ dst, const float* __restrict__ x) {
  size_t idx = (size_t)blockIdx.x * blockDim.x + threadIdx.x;
  if (idx >= (size_t)B_ * S_ * I_) return;
  int i = (int)(idx & (I_ - 1));
  size_t row = idx >> 8;           // I_=256
  int b = (int)(row & (B_ - 1));
  int t = (int)(row >> 5);
  dst[idx] = (h16)x[((size_t)b * S_ + t) * I_ + i];
}

// W (din,dout) f32 row-major -> WMMA B-operand packed f16.
// packed element idx: e=idx&15, lane=(idx>>4)&31, tile=idx>>9, nblk=tile%NB,
// kblk=tile/NB. lane: n=lane&15, half=lane>>4; koff = half*8 + (e<8? e : e+8).
__global__ void pack_w_kernel(h16* __restrict__ dst, const float* __restrict__ src,
                              int din, int dout) {
  int idx = blockIdx.x * blockDim.x + threadIdx.x;
  int total = din * dout;
  if (idx >= total) return;
  int e    = idx & 15;
  int lane = (idx >> 4) & 31;
  int tile = idx >> 9;
  int NB   = dout >> 4;
  int nblk = tile % NB;
  int kblk = tile / NB;
  int n    = lane & 15;
  int half = lane >> 4;
  int koff = half * 8 + (e < 8 ? e : e + 8);
  int k    = kblk * 32 + koff;
  int col  = nblk * 16 + n;
  dst[idx] = (h16)src[(size_t)k * dout + col];
}

// h0 (B,L,H) f32 -> hbuf f32 [32x512], hhalf f16 [32x512] for one layer
__global__ void init_h_kernel(float* __restrict__ hbuf, h16* __restrict__ hhalf,
                              const float* __restrict__ h0, int layer) {
  int idx = blockIdx.x * blockDim.x + threadIdx.x;
  if (idx >= B_ * H_) return;
  int b = idx >> 9;
  int h = idx & (H_ - 1);
  float v = h0[((size_t)b * L_ + layer) * H_ + h];
  hbuf[idx]  = v;
  hhalf[idx] = (h16)v;
}

// ---------------------------------------------------------------------------
// Phase B1: Xp[chunk] = seq_in @ [Wxz|Wxr|Wxg] + bias
// grid(24,64) blocks of 256 threads. Wave = 16 rows x 64 cols (4 tiles, shared A).
// Block = 8 waves stacked in M -> 128 rows x 64 cols. Cols 0..1535 (3 gates x 512).
// ---------------------------------------------------------------------------
__global__ void __launch_bounds__(256) xp_gemm_kernel(
    const h16* __restrict__ seq_in, int din,
    const h16* __restrict__ pkz, const h16* __restrict__ pkr,
    const h16* __restrict__ pkg,
    const float* __restrict__ bz, const float* __restrict__ br,
    const float* __restrict__ bg,
    float* __restrict__ xp, int t0) {
  int lane = threadIdx.x & 31;
  int w    = threadIdx.x >> 5;
  int nn   = lane & 15;
  int kb8  = (lane >> 4) * 8;

  int colbase = blockIdx.x * 64;        // 0..1535
  int gate    = colbase >> 9;
  int gcol    = colbase & (H_ - 1);     // col within gate matrix
  const h16*   pk   = (gate == 0) ? pkz : (gate == 1) ? pkr : pkg;
  const float* bias = (gate == 0) ? bz  : (gate == 1) ? br  : bg;

  int rowbase = blockIdx.y * 128 + w * 16;   // chunk-local row (0..8191)
  const h16* arow = seq_in + (size_t)(t0 * B_ + rowbase + (lane & 15)) * din;

  v8f acc[4] = {v8f{}, v8f{}, v8f{}, v8f{}};
  int nkb = din >> 5;
  int nb0 = gcol >> 4;                   // first col-tile of this wave (of 32)
  for (int kb = 0; kb < nkb; ++kb) {
    v16h a = load_a(arow, kb, kb8);
    const h16* bt = pk + ((size_t)(kb * 32 + nb0) * 512) + lane * 16;
    v16h b0 = *(const v16h*)(bt);
    v16h b1 = *(const v16h*)(bt + 512);
    v16h b2 = *(const v16h*)(bt + 1024);
    v16h b3 = *(const v16h*)(bt + 1536);
    acc[0] = wmma16(a, b0, acc[0]);
    acc[1] = wmma16(a, b1, acc[1]);
    acc[2] = wmma16(a, b2, acc[2]);
    acc[3] = wmma16(a, b3, acc[3]);
  }
#pragma unroll
  for (int nb = 0; nb < 4; ++nb) {
#pragma unroll
    for (int vv = 0; vv < 8; ++vv) {
      int m   = vv + ((lane >> 4) << 3);
      int row = rowbase + m;
      int col = colbase + nb * 16 + nn;
      xp[(size_t)row * 1536 + col] = acc[nb][vv] + bias[gcol + nb * 16 + nn];
    }
  }
}

// ---------------------------------------------------------------------------
// Grid barrier (sense via monotonically increasing generation counter).
// Fences executed by all threads so every CU's L0 is released/invalidated.
// ---------------------------------------------------------------------------
__device__ __forceinline__ void grid_barrier(unsigned* cnt, unsigned* gen) {
  __syncthreads();
  __threadfence();
  if (threadIdx.x == 0) {
    unsigned g   = __atomic_load_n(gen, __ATOMIC_RELAXED);
    unsigned old = atomicAdd(cnt, 1u);
    if (old == NWG_REC - 1) {
      __atomic_store_n(cnt, 0u, __ATOMIC_RELAXED);
      __atomic_fetch_add(gen, 1u, __ATOMIC_RELEASE);
    } else {
      while (__atomic_load_n(gen, __ATOMIC_ACQUIRE) == g) {
        __builtin_amdgcn_s_sleep(1);
      }
    }
  }
  __syncthreads();
  __threadfence();
}

// ---------------------------------------------------------------------------
// Phase B2: persistent recurrent kernel. 8 WGs x 256 thr. Each WG owns a
// 64-column slice of h; wave w -> tile (mblk=w&1, nblk=w>>1). z and hp live
// in registers across the intra-step barrier.
// ---------------------------------------------------------------------------
__global__ void __launch_bounds__(256) recur_kernel(
    const float* __restrict__ xp,        // CT*32 x 1536 (z|r|g)
    const h16* __restrict__ pkwhz, const h16* __restrict__ pkwhr,
    const h16* __restrict__ pkwhg,
    float* __restrict__ hbuf,            // 32x512 f32
    h16* __restrict__ hhalf,             // 32x512 f16 (A operand for stage1)
    h16* __restrict__ rhbuf,             // 32x512 f16 (A operand for stage2)
    h16* __restrict__ seq_out,           // 65536x512 f16
    float* __restrict__ out_hidden,      // (B,L,H) slice of d_out
    unsigned* __restrict__ bar_cnt, unsigned* __restrict__ bar_gen,
    int t0, int nsteps, int layer) {
  int lane = threadIdx.x & 31;
  int w    = threadIdx.x >> 5;
  int mblk = w & 1;
  int nblk = w >> 1;
  int nn   = lane & 15;
  int kb8  = (lane >> 4) * 8;
  int colbase = blockIdx.x * 64;
  int tilecol = colbase + nblk * 16;     // 0..511
  int nbg     = tilecol >> 4;            // global col-tile (0..31)
  int arow    = (mblk * 16 + (lane & 15)) * H_;   // A row offset (row-major, 512)

  for (int step = 0; step < nsteps; ++step) {
    int t = t0 + step;
    // -------- stage 1: z, r --------
    v8f az, ar;
    float hp[8];
#pragma unroll
    for (int vv = 0; vv < 8; ++vv) {
      int m = mblk * 16 + vv + ((lane >> 4) << 3);
      size_t xr = (size_t)(step * B_ + m) * 1536;
      az[vv] = xp[xr + 0   + tilecol + nn];
      ar[vv] = xp[xr + 512 + tilecol + nn];
      hp[vv] = hbuf[m * H_ + tilecol + nn];
    }
    const h16* ap = hhalf + arow;
#pragma unroll 4
    for (int kb = 0; kb < 16; ++kb) {
      v16h a  = load_a(ap, kb, kb8);
      size_t bo_ = ((size_t)(kb * 32 + nbg) * 512) + lane * 16;
      v16h bz_ = *(const v16h*)(pkwhz + bo_);
      v16h br_ = *(const v16h*)(pkwhr + bo_);
      az = wmma16(a, bz_, az);
      ar = wmma16(a, br_, ar);
    }
    float zreg[8];
#pragma unroll
    for (int vv = 0; vv < 8; ++vv) {
      int m = mblk * 16 + vv + ((lane >> 4) << 3);
      float z = 1.0f / (1.0f + __expf(-az[vv]));
      float r = 1.0f / (1.0f + __expf(-ar[vv]));
      zreg[vv] = z;
      rhbuf[m * H_ + tilecol + nn] = (h16)(r * hp[vv]);
    }
    grid_barrier(bar_cnt, bar_gen);

    // -------- stage 2: g, h --------
    v8f ag;
#pragma unroll
    for (int vv = 0; vv < 8; ++vv) {
      int m = mblk * 16 + vv + ((lane >> 4) << 3);
      ag[vv] = xp[(size_t)(step * B_ + m) * 1536 + 1024 + tilecol + nn];
    }
    const h16* ap2 = rhbuf + arow;
#pragma unroll 4
    for (int kb = 0; kb < 16; ++kb) {
      v16h a  = load_a(ap2, kb, kb8);
      v16h bg_ = *(const v16h*)(pkwhg + ((size_t)(kb * 32 + nbg) * 512) + lane * 16);
      ag = wmma16(a, bg_, ag);
    }
#pragma unroll
    for (int vv = 0; vv < 8; ++vv) {
      int m   = mblk * 16 + vv + ((lane >> 4) << 3);
      int col = tilecol + nn;
      float g = tanhf(ag[vv]);
      float h = zreg[vv] * hp[vv] + (1.0f - zreg[vv]) * g;
      hbuf[m * H_ + col]  = h;
      hhalf[m * H_ + col] = (h16)h;
      seq_out[((size_t)t * B_ + m) * H_ + col] = (h16)h;
      if (t == S_ - 1) out_hidden[((size_t)m * L_ + layer) * H_ + col] = h;
    }
    grid_barrier(bar_cnt, bar_gen);
  }
}

// ---------------------------------------------------------------------------
// Phase C: layer_output = seq_l2 @ Wo + bo. rows=65536, cols=256, K=512.
// grid(4,512) blocks of 256 threads; wave = 16 rows x 64 cols.
// ---------------------------------------------------------------------------
__global__ void __launch_bounds__(256) out_gemm_kernel(
    const h16* __restrict__ seq_in, const h16* __restrict__ pkwo,
    const float* __restrict__ bo, float* __restrict__ out) {
  int lane = threadIdx.x & 31;
  int w    = threadIdx.x >> 5;
  int nn   = lane & 15;
  int kb8  = (lane >> 4) * 8;
  int colbase = blockIdx.x * 64;             // 0..255
  int rowbase = blockIdx.y * 128 + w * 16;   // 0..65535
  const h16* arow = seq_in + (size_t)(rowbase + (lane & 15)) * H_;

  v8f acc[4] = {v8f{}, v8f{}, v8f{}, v8f{}};
  int nb0 = colbase >> 4;
  for (int kb = 0; kb < 16; ++kb) {
    v16h a = load_a(arow, kb, kb8);
    const h16* bt = pkwo + ((size_t)(kb * 16 + nb0) * 512) + lane * 16;
    v16h b0 = *(const v16h*)(bt);
    v16h b1 = *(const v16h*)(bt + 512);
    v16h b2 = *(const v16h*)(bt + 1024);
    v16h b3 = *(const v16h*)(bt + 1536);
    acc[0] = wmma16(a, b0, acc[0]);
    acc[1] = wmma16(a, b1, acc[1]);
    acc[2] = wmma16(a, b2, acc[2]);
    acc[3] = wmma16(a, b3, acc[3]);
  }
#pragma unroll
  for (int nb = 0; nb < 4; ++nb) {
#pragma unroll
    for (int vv = 0; vv < 8; ++vv) {
      int m = vv + ((lane >> 4) << 3);
      int r = rowbase + m;                 // = t*32+b
      int b = r & (B_ - 1);
      int t = r >> 5;
      int col = colbase + nb * 16 + nn;
      out[((size_t)b * S_ + t) * O_ + col] = acc[nb][vv] + bo[col];
    }
  }
}

// ---------------------------------------------------------------------------
// Host driver
// ---------------------------------------------------------------------------
extern "C" void kernel_launch(void* const* d_in, const int* in_sizes, int n_in,
                              void* d_out, int out_size, void* d_ws, size_t ws_size,
                              hipStream_t stream) {
  (void)in_sizes; (void)n_in; (void)out_size; (void)ws_size;
  const float* x  = (const float*)d_in[0];
  const float* h0 = (const float*)d_in[1];
  // params per layer l (base 2+9l): 0:Wxz 1:bxz 2:Wxr 3:bxr 4:Wxg 5:bxg 6:Whz 7:Whr 8:Whg
  // then 29:Wo 30:bo

  char*  ws  = (char*)d_ws;
  size_t cur = 0;
  auto alloc = [&](size_t bytes) {
    size_t o = cur;
    cur += (bytes + 255) & ~(size_t)255;
    return o;
  };

  size_t pk_wx[L_][3], pk_wh[L_][3];
  for (int l = 0; l < L_; ++l)
    for (int g = 0; g < 3; ++g)
      pk_wx[l][g] = alloc((size_t)(l == 0 ? I_ : H_) * H_ * sizeof(h16));
  for (int l = 0; l < L_; ++l)
    for (int g = 0; g < 3; ++g)
      pk_wh[l][g] = alloc((size_t)H_ * H_ * sizeof(h16));
  size_t pk_wo  = alloc((size_t)H_ * O_ * sizeof(h16));
  size_t xbf_o  = alloc((size_t)B_ * S_ * I_ * sizeof(h16));
  size_t seqA_o = alloc((size_t)B_ * S_ * H_ * sizeof(h16));
  size_t seqB_o = alloc((size_t)B_ * S_ * H_ * sizeof(h16));
  size_t xp_o   = alloc((size_t)CT * B_ * 1536 * sizeof(float));
  size_t hbuf_o = alloc((size_t)B_ * H_ * sizeof(float));
  size_t hhalf_o= alloc((size_t)B_ * H_ * sizeof(h16));
  size_t rh_o   = alloc((size_t)B_ * H_ * sizeof(h16));
  size_t bar_o  = alloc(256);

  unsigned* bar_cnt = (unsigned*)(ws + bar_o);
  unsigned* bar_gen = (unsigned*)(ws + bar_o) + 1;
  hipMemsetAsync(ws + bar_o, 0, 256, stream);

  // Phase A: pack x and weights to f16 WMMA layouts.
  pack_x_kernel<<<(B_ * S_ * I_) / 256, 256, 0, stream>>>((h16*)(ws + xbf_o), x);
  for (int l = 0; l < L_; ++l) {
    int din = (l == 0) ? I_ : H_;
    int base = 2 + 9 * l;
    const float* Wx[3] = {(const float*)d_in[base + 0], (const float*)d_in[base + 2],
                          (const float*)d_in[base + 4]};
    const float* Wh[3] = {(const float*)d_in[base + 6], (const float*)d_in[base + 7],
                          (const float*)d_in[base + 8]};
    for (int g = 0; g < 3; ++g) {
      pack_w_kernel<<<(din * H_ + 255) / 256, 256, 0, stream>>>(
          (h16*)(ws + pk_wx[l][g]), Wx[g], din, H_);
      pack_w_kernel<<<(H_ * H_ + 255) / 256, 256, 0, stream>>>(
          (h16*)(ws + pk_wh[l][g]), Wh[g], H_, H_);
    }
  }
  pack_w_kernel<<<(H_ * O_ + 255) / 256, 256, 0, stream>>>(
      (h16*)(ws + pk_wo), (const float*)d_in[29], H_, O_);

  float* out        = (float*)d_out;
  float* out_hidden = out + (size_t)B_ * S_ * O_;

  // Per-layer: chunked Xp GEMM + persistent recurrence.
  for (int l = 0; l < L_; ++l) {
    int din = (l == 0) ? I_ : H_;
    int base = 2 + 9 * l;
    const h16* seq_in = (l == 0) ? (const h16*)(ws + xbf_o)
                     : (l == 1) ? (const h16*)(ws + seqA_o)
                                : (const h16*)(ws + seqB_o);
    h16* seq_out = (l == 0) ? (h16*)(ws + seqA_o)
                 : (l == 1) ? (h16*)(ws + seqB_o)
                            : (h16*)(ws + seqA_o);
    const float* bz = (const float*)d_in[base + 1];
    const float* br = (const float*)d_in[base + 3];
    const float* bg = (const float*)d_in[base + 5];

    init_h_kernel<<<(B_ * H_) / 256, 256, 0, stream>>>(
        (float*)(ws + hbuf_o), (h16*)(ws + hhalf_o), h0, l);

    for (int c = 0; c < S_ / CT; ++c) {
      int t0 = c * CT;
      xp_gemm_kernel<<<dim3(24, 64), 256, 0, stream>>>(
          seq_in, din,
          (const h16*)(ws + pk_wx[l][0]), (const h16*)(ws + pk_wx[l][1]),
          (const h16*)(ws + pk_wx[l][2]),
          bz, br, bg, (float*)(ws + xp_o), t0);
      recur_kernel<<<NWG_REC, 256, 0, stream>>>(
          (const float*)(ws + xp_o),
          (const h16*)(ws + pk_wh[l][0]), (const h16*)(ws + pk_wh[l][1]),
          (const h16*)(ws + pk_wh[l][2]),
          (float*)(ws + hbuf_o), (h16*)(ws + hhalf_o), (h16*)(ws + rh_o),
          seq_out, out_hidden, bar_cnt, bar_gen, t0, CT, l);
    }
  }

  // Phase C: output projection (layer 2 output sequence lives in seqA).
  out_gemm_kernel<<<dim3(4, 512), 256, 0, stream>>>(
      (const h16*)(ws + seqA_o), (const h16*)(ws + pk_wo),
      (const float*)d_in[30], out);
}
